// ASConv2d_28870770163904
// MI455X (gfx1250) — compile-verified, matrix operand order checked by reading, not run on hardware
//
#include <hip/hip_runtime.h>
#include <hip/hip_bf16.h>

typedef __attribute__((ext_vector_type(16))) _Float16 v16h;
typedef __attribute__((ext_vector_type(8)))  _Float16 v8h;
typedef __attribute__((ext_vector_type(8)))  float    v8f;

#define BATCH 2
#define INC   32
#define OUTC  64
#define H     192
#define W     192
#define NPT   9           // 3x3 sampling points
#define KDIM  (NPT * INC) // 288, ordered k = n*32 + c
#define HW    (H * W)     // 36864
#define NPIX  (BATCH * HW)
#define KSTEPS (KDIM / 32) // 9

// ---------------------------------------------------------------------------
// Kernel 1: per-pixel scale map A (1-ch 3x3 conv) + bilinear 9-point gather.
// Emits Xoff[pix][k] (f16), k = n*32 + c.  One thread per output pixel.
// ---------------------------------------------------------------------------
__global__ void asconv_gather(const float* __restrict__ x,
                              const float* __restrict__ p1w,
                              const float* __restrict__ p1b,
                              _Float16* __restrict__ xoff) {
    int pix = blockIdx.x * blockDim.x + threadIdx.x;
    if (pix >= NPIX) return;
    int b   = pix / HW;
    int rem = pix - b * HW;
    int i = rem / W;
    int j = rem - i * W;

    const float* xb = x + (size_t)b * INC * HW;

    // A = conv3x3(x, p1_w) + p1_b  (zero padding)
    float A = p1b[0];
    for (int c = 0; c < INC; ++c) {
        const float* xc = xb + c * HW;
        const float* wc = p1w + c * 9;
#pragma unroll
        for (int t = 0; t < 9; ++t) {
            int yy = i + (t / 3) - 1;
            int xx = j + (t % 3) - 1;
            if (yy >= 0 && yy < H && xx >= 0 && xx < W)
                A = fmaf(xc[yy * W + xx], wc[t], A);
        }
    }

    _Float16* dstrow = xoff + (size_t)pix * KDIM;

#pragma unroll
    for (int n = 0; n < NPT; ++n) {
        const float pnx = (float)(n / 3 - 1);
        const float pny = (float)(n % 3 - 1);
        // padded coords: Hp = Wp = 194, center at (i+1, j+1)
        float px = (float)(i + 1) + A * pnx;
        float py = (float)(j + 1) + A * pny;

        float flx = floorf(px), fly = floorf(py);
        float fx_lt = fminf(fmaxf(flx,        0.f), 193.f);
        float fy_lt = fminf(fmaxf(fly,        0.f), 193.f);
        float fx_rb = fminf(fmaxf(flx + 1.f,  0.f), 193.f);
        float fy_rb = fminf(fmaxf(fly + 1.f,  0.f), 193.f);
        float pxc   = fminf(fmaxf(px, 0.f), 193.f);
        float pyc   = fminf(fmaxf(py, 0.f), 193.f);

        float wxl = 1.f + (fx_lt - pxc);   // x-weight for lt row
        float wxr = 1.f - (fx_rb - pxc);   // x-weight for rb row
        float wyl = 1.f + (fy_lt - pyc);
        float wyr = 1.f - (fy_rb - pyc);
        float g_lt = wxl * wyl, g_rb = wxr * wyr;
        float g_lb = wxl * wyr, g_rt = wxr * wyl;

        int qxl = (int)fx_lt, qyl = (int)fy_lt;
        int qxr = (int)fx_rb, qyr = (int)fy_rb;
        // padded map: interior (1..192) maps to x[q-1]; border is zero
        bool vxl = (qxl >= 1) && (qxl <= H);
        bool vxr = (qxr >= 1) && (qxr <= H);
        bool vyl = (qyl >= 1) && (qyl <= W);
        bool vyr = (qyr >= 1) && (qyr <= W);
        int o_lt = (qxl - 1) * W + (qyl - 1);
        int o_rb = (qxr - 1) * W + (qyr - 1);
        int o_lb = (qxl - 1) * W + (qyr - 1);
        int o_rt = (qxr - 1) * W + (qyl - 1);
        bool v_lt = vxl && vyl, v_rb = vxr && vyr;
        bool v_lb = vxl && vyr, v_rt = vxr && vyl;

        _Float16* dst = dstrow + n * INC;
        for (int c = 0; c < INC; ++c) {
            const float* xc = xb + c * HW;
            float a0 = v_lt ? xc[o_lt] : 0.f;
            float a1 = v_rb ? xc[o_rb] : 0.f;
            float a2 = v_lb ? xc[o_lb] : 0.f;
            float a3 = v_rt ? xc[o_rt] : 0.f;
            float val = g_lt * a0 + g_rb * a1 + g_lb * a2 + g_rt * a3;
            dst[c] = (_Float16)val;
        }
    }
}

// ---------------------------------------------------------------------------
// Kernel 2: pack conv_w (64,32,3,3) into WMMA B-fragment order (f16).
// Bfrag[((s*4 + t)*32 + lane)*16 + h]; lane holds column o = t*16 + lane%16,
// K striping identical to the A-matrix layout (kb = lane<16 ? 0 : 8).
// k = n*32 + c  ->  weight = conv_w[o][c][n].
// ---------------------------------------------------------------------------
__global__ void asconv_packB(const float* __restrict__ convw,
                             _Float16* __restrict__ bfrag) {
    int e = blockIdx.x * blockDim.x + threadIdx.x;
    const int total = KSTEPS * 4 * 32 * 16;
    if (e >= total) return;
    int h    = e & 15;
    int lane = (e >> 4) & 31;
    int t    = (e >> 9) & 3;
    int s    = e >> 11;

    int o  = t * 16 + (lane & 15);
    int kb = (lane < 16) ? 0 : 8;
    int k  = s * 32 + kb + h + ((h < 8) ? 0 : 8); // halves 8..15 -> K+16 block
    int n  = k >> 5;       // k / 32
    int c  = k & 31;       // k % 32
    bfrag[e] = (_Float16)convw[(o * INC + c) * NPT + n];
}

// ---------------------------------------------------------------------------
// Kernel 3: WMMA GEMM.  M = NPIX (pixels), N = 64 (out ch), K = 288.
// One wave per 16-pixel tile; 4 f32 accumulators (N tiles); 9 K-steps.
// ---------------------------------------------------------------------------
__global__ void asconv_wmma_gemm(const _Float16* __restrict__ xoff,
                                 const _Float16* __restrict__ bfrag,
                                 float* __restrict__ out) {
    int gtid = blockIdx.x * blockDim.x + threadIdx.x;
    int tile = gtid >> 5;                 // wave id == pixel tile
    int lane = threadIdx.x & 31;
    if (tile >= NPIX / 16) return;

    int m   = lane & 15;
    int kb  = (lane < 16) ? 0 : 8;
    const _Float16* arow = xoff + (size_t)(tile * 16 + m) * KDIM;

    v8f acc0 = {}, acc1 = {}, acc2 = {}, acc3 = {};

#pragma unroll
    for (int s = 0; s < KSTEPS; ++s) {
        union { v16h v; struct { v8h lo, hi; } p; } ua;
        ua.p.lo = *(const v8h*)(arow + s * 32 + kb);        // K = kb..kb+7
        ua.p.hi = *(const v8h*)(arow + s * 32 + kb + 16);   // K = kb+16..kb+23
        v16h a = ua.v;

        const _Float16* bs = bfrag + (size_t)(s * 4) * 32 * 16 + lane * 16;
        v16h b0 = *(const v16h*)(bs);
        v16h b1 = *(const v16h*)(bs + 32 * 16);
        v16h b2 = *(const v16h*)(bs + 2 * 32 * 16);
        v16h b3 = *(const v16h*)(bs + 3 * 32 * 16);

        acc0 = __builtin_amdgcn_wmma_f32_16x16x32_f16(false, a, false, b0,
                                                      (short)0, acc0, false, false);
        acc1 = __builtin_amdgcn_wmma_f32_16x16x32_f16(false, a, false, b1,
                                                      (short)0, acc1, false, false);
        acc2 = __builtin_amdgcn_wmma_f32_16x16x32_f16(false, a, false, b2,
                                                      (short)0, acc2, false, false);
        acc3 = __builtin_amdgcn_wmma_f32_16x16x32_f16(false, a, false, b3,
                                                      (short)0, acc3, false, false);
    }

    // D layout: lane = column (o), VGPR r = row m = r + (lane<16?0:8).
    // Rows are consecutive pixels -> two contiguous float4 stores per acc.
    int pix0 = tile * 16 + ((lane < 16) ? 0 : 8);
    int b    = pix0 / HW;            // tile never straddles batch (HW%16==0)
    int rem  = pix0 - b * HW;
    int ocol = lane & 15;

#pragma unroll
    for (int t = 0; t < 4; ++t) {
        v8f acc = (t == 0) ? acc0 : (t == 1) ? acc1 : (t == 2) ? acc2 : acc3;
        int o = t * 16 + ocol;
        float* dst = out + (size_t)b * OUTC * HW + (size_t)o * HW + rem;
        float4 lo = make_float4(acc[0], acc[1], acc[2], acc[3]);
        float4 hi = make_float4(acc[4], acc[5], acc[6], acc[7]);
        *(float4*)(dst)     = lo;
        *(float4*)(dst + 4) = hi;
    }
}

// ---------------------------------------------------------------------------
// Launch
// inputs: 0:x 1:conv_w 2:p_conv_w(unused) 3:p_conv_b(unused) 4:p1_w 5:p1_b
// ws: [0, 36KB)  Bfrag ; [64KB, 64KB+42.5MB) Xoff f16
// ---------------------------------------------------------------------------
extern "C" void kernel_launch(void* const* d_in, const int* in_sizes, int n_in,
                              void* d_out, int out_size, void* d_ws, size_t ws_size,
                              hipStream_t stream) {
    const float* x     = (const float*)d_in[0];
    const float* convw = (const float*)d_in[1];
    const float* p1w   = (const float*)d_in[4];
    const float* p1b   = (const float*)d_in[5];
    float* out         = (float*)d_out;

    _Float16* bfrag = (_Float16*)d_ws;                              // 36,864 B
    _Float16* xoff  = (_Float16*)((char*)d_ws + (64 * 1024));       // 42.5 MB

    // pack weights into fragment order (tiny)
    {
        const int total = KSTEPS * 4 * 32 * 16;   // 18432
        asconv_packB<<<(total + 255) / 256, 256, 0, stream>>>(convw, bfrag);
    }
    // gather / bilinear -> f16 activation matrix
    {
        const int total = NPIX;                   // 73728
        asconv_gather<<<(total + 255) / 256, 256, 0, stream>>>(x, p1w, p1b, xoff);
    }
    // WMMA GEMM: one wave per 16-pixel tile, 4608 waves
    {
        const int waves   = NPIX / 16;            // 4608
        const int threads = waves * 32;           // 147456
        asconv_wmma_gemm<<<(threads + 255) / 256, 256, 0, stream>>>(xoff, bfrag, out);
    }
}
